// VQVAEVC_6390911336901
// MI455X (gfx1250) — compile-verified, matrix operand order checked by reading, not run on hardware
//
#include <hip/hip_runtime.h>

// ---------------------------------------------------------------------------
// VQ-VAE forward for MI455X (gfx1250, wave32).
// Conv / convT layers and the VQ distance GEMM run on v_wmma_f32_16x16x32_bf16
// (bf16 in, f32 accumulate). Activations are bf16 channel-last. Each block
// stages its activation window into LDS with async global->LDS copies
// (ASYNCcnt) and computes a 256-out-channel x 128-position macro-tile; each
// wave reuses A-fragments across 8 N-tiles to keep L2 weight traffic low.
// ---------------------------------------------------------------------------

typedef __attribute__((ext_vector_type(16))) __bf16 v16bf;
typedef __attribute__((ext_vector_type(8)))  __bf16 v8bf;
typedef __attribute__((ext_vector_type(8)))  float  v8f;

// exact pointee type expected by the async-LDS builtin (per hipcc diagnostic):
typedef int vi4 __attribute__((vector_size(16)));
typedef __attribute__((address_space(1))) vi4 gvi4;   // global (prints __device__)
typedef __attribute__((address_space(3))) vi4 lvi4;   // LDS

__device__ __forceinline__ v8bf zero8bf() {
  v8bf r;
#pragma unroll
  for (int i = 0; i < 8; ++i) r[i] = (__bf16)0.0f;
  return r;
}

union FragBF { v16bf v; v8bf h[2]; };

// ---- async global->LDS staging (CDNA5 ASYNCcnt path, with sync fallback) ---
#if defined(__has_builtin)
#if __has_builtin(__builtin_amdgcn_global_load_async_to_lds_b128)
#define HAVE_ASYNC_LDS 1
#endif
#if __has_builtin(__builtin_amdgcn_s_wait_asynccnt)
#define HAVE_WAIT_ASYNC 1
#endif
#endif

__device__ __forceinline__ void stage16(__bf16* dst_lds, const __bf16* src_g) {
#ifdef HAVE_ASYNC_LDS
  __builtin_amdgcn_global_load_async_to_lds_b128(
      (gvi4*)(void*)src_g, (lvi4*)dst_lds, 0, 0);
#else
  *(v8bf*)dst_lds = *(const v8bf*)src_g;
#endif
}

__device__ __forceinline__ void wait_async_lds() {
#ifdef HAVE_WAIT_ASYNC
  __builtin_amdgcn_s_wait_asynccnt(0);
#elif defined(HAVE_ASYNC_LDS)
  asm volatile("s_wait_asynccnt 0" ::: "memory");
#endif
}

// ---------------------------------------------------------------------------
// conv1d (k=3, pad=1, stride STRIDE) as implicit GEMM with LDS-staged input.
// X: [B, Lin, CIN] bf16, W: [COUT, 3*CIN] bf16 ([o][tap][c]), Y: [B,Lout,COUT].
// Block: 8 waves, COUT channels x BLOCK_N positions. Wave: MREP m-tiles x
// NREP n-tiles; A-fragments (global/L2) reused NREP x, B-fragments from LDS.
// ---------------------------------------------------------------------------
template <int CIN, int COUT, int STRIDE, bool RELU, int MREP, int NREP>
__global__ __launch_bounds__(256) void conv_wmma(
    const __bf16* __restrict__ X, const __bf16* __restrict__ W,
    const float* __restrict__ bias, __bf16* __restrict__ Y,
    int Lin, int Lout)
{
  constexpr int KDIM    = 3 * CIN;
  constexpr int MT      = COUT / 16;
  constexpr int MG      = MT / MREP;          // m-groups of waves
  constexpr int NG      = 8 / MG;             // n-groups of waves
  constexpr int BLOCK_N = NG * NREP * 16;
  constexpr int NROWS   = STRIDE * (BLOCK_N - 1) + 3;
  constexpr int PITCH   = CIN + 8;            // bank-spread pad (16B aligned)
  constexpr int CPR     = CIN / 8;            // 16B chunks per input row
  static_assert(MG * NG == 8, "wave decomposition");

  __shared__ __bf16 sx[NROWS * PITCH];

  const int tid  = threadIdx.x;
  const int wave = tid >> 5;
  const int lane = tid & 31;
  const int l    = lane & 15;
  const int half = lane >> 4;
  const int mg   = wave % MG;
  const int ngr  = wave / MG;
  const int m0   = mg * (MREP * 16);
  const int nloc0 = ngr * (NREP * 16);

  const long nblk = (long)blockIdx.x * BLOCK_N;     // flattened (b, t) base
  const int  b    = (int)(nblk / Lout);
  const int  tblk = (int)(nblk % Lout);             // BLOCK_N | Lout
  const __bf16* xb = X + (size_t)b * Lin * CIN;

  // ---- stage activation window [tblk*S-1, tblk*S-1+NROWS) into LDS ----
  {
    const int base = tblk * STRIDE - 1;
    for (int i = tid; i < NROWS * CPR; i += 256) {
      int r  = i / CPR;
      int c  = (i - r * CPR) * 8;
      int sp = base + r;
      __bf16* dst = &sx[r * PITCH + c];
      if (sp >= 0 && sp < Lin) stage16(dst, xb + (size_t)sp * CIN + c);
      else                     *(v8bf*)dst = zero8bf();
    }
    wait_async_lds();
  }
  __syncthreads();

  v8f acc[MREP][NREP] = {};

  for (int kk = 0; kk < KDIM; kk += 32) {
    // A fragments (weights, L2-resident), reused across NREP n-tiles
    FragBF fa[MREP];
#pragma unroll
    for (int mi = 0; mi < MREP; ++mi) {
      const __bf16* arow = W + (size_t)(m0 + mi * 16 + l) * KDIM + kk + half * 8;
      fa[mi].h[0] = *(const v8bf*)(arow);
      fa[mi].h[1] = *(const v8bf*)(arow + 16);
    }
    const int K0  = kk + half * 8;
    const int tp0 = K0 / CIN,        c0 = K0 - tp0 * CIN;
    const int K1  = kk + 16 + half * 8;
    const int tp1 = K1 / CIN,        c1 = K1 - tp1 * CIN;
#pragma unroll
    for (int nr = 0; nr < NREP; ++nr) {
      const int tl = nloc0 + nr * 16 + l;           // position within block
      FragBF fb;
      fb.h[0] = *(const v8bf*)(&sx[(tl * STRIDE + tp0) * PITCH + c0]);
      fb.h[1] = *(const v8bf*)(&sx[(tl * STRIDE + tp1) * PITCH + c1]);
#pragma unroll
      for (int mi = 0; mi < MREP; ++mi)
        acc[mi][nr] = __builtin_amdgcn_wmma_f32_16x16x32_bf16(
            false, fa[mi].v, false, fb.v, (short)0, acc[mi][nr], false, false);
    }
  }

#pragma unroll
  for (int nr = 0; nr < NREP; ++nr) {
    const int tl = nloc0 + nr * 16 + l;
    __bf16* yrow = Y + ((size_t)b * Lout + tblk + tl) * COUT;
#pragma unroll
    for (int mi = 0; mi < MREP; ++mi) {
      const int obase = m0 + mi * 16 + half * 8;
      v8bf ov;
#pragma unroll
      for (int r = 0; r < 8; ++r) {
        float v = acc[mi][nr][r] + bias[obase + r];
        if (RELU) v = fmaxf(v, 0.0f);
        ov[r] = (__bf16)v;
      }
      *(v8bf*)(yrow + obase) = ov;
    }
  }
}

// ---------------------------------------------------------------------------
// Transposed conv1d (k=3, stride 2, pad 1, output_padding 1), 256->256,
// split by output parity. Even t: tap k=1 (K=256). Odd t: k=0 -> x[u],
// k=2 -> x[u+1] (K=512). LDS-staged input window, same wave decomposition.
// FINAL writes fp32 channel-first [B, 256, 2*Lin], no ReLU.
// ---------------------------------------------------------------------------
template <bool ODD, bool FINAL>
__global__ __launch_bounds__(256) void convT_wmma(
    const __bf16* __restrict__ X, const __bf16* __restrict__ W,
    const float* __restrict__ bias, void* __restrict__ Yv, int Lin)
{
  constexpr int C       = 256;
  constexpr int KDIM    = ODD ? 512 : 256;
  constexpr int MREP    = 2;
  constexpr int NREP    = 8;
  constexpr int BLOCK_N = 128;                 // u-positions per block
  constexpr int NROWS   = ODD ? (BLOCK_N + 1) : BLOCK_N;
  constexpr int PITCH   = C + 8;
  constexpr int CPR     = C / 8;

  __shared__ __bf16 sx[NROWS * PITCH];

  const int tid  = threadIdx.x;
  const int wave = tid >> 5;
  const int lane = tid & 31;
  const int l    = lane & 15;
  const int half = lane >> 4;
  const int m0   = wave * (MREP * 16);         // 8 waves x 32 ch = 256
  const int Lout = 2 * Lin;

  const long nblk = (long)blockIdx.x * BLOCK_N;     // flattened (b, u) base
  const int  b    = (int)(nblk / Lin);
  const int  ublk = (int)(nblk % Lin);
  const __bf16* xb = X + (size_t)b * Lin * C;

  {
    for (int i = tid; i < NROWS * CPR; i += 256) {
      int r  = i / CPR;
      int c  = (i - r * CPR) * 8;
      int sp = ublk + r;
      __bf16* dst = &sx[r * PITCH + c];
      if (sp < Lin) stage16(dst, xb + (size_t)sp * C + c);
      else          *(v8bf*)dst = zero8bf();
    }
    wait_async_lds();
  }
  __syncthreads();

  v8f acc[MREP][NREP] = {};

  for (int kk = 0; kk < KDIM; kk += 32) {
    FragBF fa[MREP];
#pragma unroll
    for (int mi = 0; mi < MREP; ++mi) {
      const __bf16* arow = W + (size_t)(m0 + mi * 16 + l) * KDIM + kk + half * 8;
      fa[mi].h[0] = *(const v8bf*)(arow);
      fa[mi].h[1] = *(const v8bf*)(arow + 16);
    }
    const int K0 = kk + half * 8;
    const int K1 = kk + 16 + half * 8;
    const int tp0 = ODD ? (K0 >> 8) : 0, c0 = ODD ? (K0 & 255) : K0;
    const int tp1 = ODD ? (K1 >> 8) : 0, c1 = ODD ? (K1 & 255) : K1;
#pragma unroll
    for (int nr = 0; nr < NREP; ++nr) {
      const int ul = nr * 16 + l;
      FragBF fb;
      fb.h[0] = *(const v8bf*)(&sx[(ul + tp0) * PITCH + c0]);
      fb.h[1] = *(const v8bf*)(&sx[(ul + tp1) * PITCH + c1]);
#pragma unroll
      for (int mi = 0; mi < MREP; ++mi)
        acc[mi][nr] = __builtin_amdgcn_wmma_f32_16x16x32_bf16(
            false, fa[mi].v, false, fb.v, (short)0, acc[mi][nr], false, false);
    }
  }

#pragma unroll
  for (int nr = 0; nr < NREP; ++nr) {
    const int ul = nr * 16 + l;
    const int t  = 2 * (ublk + ul) + (ODD ? 1 : 0);
    if (FINAL) {
      float* Y = (float*)Yv;                   // [B, 256, Lout] fp32, no ReLU
#pragma unroll
      for (int mi = 0; mi < MREP; ++mi) {
        const int obase = m0 + mi * 16 + half * 8;
#pragma unroll
        for (int r = 0; r < 8; ++r)
          Y[((size_t)b * C + obase + r) * Lout + t] = acc[mi][nr][r] + bias[obase + r];
      }
    } else {
      __bf16* Y = (__bf16*)Yv;
      __bf16* yrow = Y + ((size_t)b * Lout + t) * C;
#pragma unroll
      for (int mi = 0; mi < MREP; ++mi) {
        const int obase = m0 + mi * 16 + half * 8;
        v8bf ov;
#pragma unroll
        for (int r = 0; r < 8; ++r)
          ov[r] = (__bf16)fmaxf(acc[mi][nr][r] + bias[obase + r], 0.0f);
        *(v8bf*)(yrow + obase) = ov;
      }
    }
  }
}

// ---------------------------------------------------------------------------
// VQ: scores = CB[512,64] . Z[n,64]^T via WMMA, fused running argmin of
// ||c||^2 - 2 z.c. One wave per 16 z-vectors.
// ---------------------------------------------------------------------------
__global__ __launch_bounds__(256) void vq_argmin(
    const __bf16* __restrict__ Z, const __bf16* __restrict__ CBB,
    const float* __restrict__ cnorm, int* __restrict__ idx)
{
  const int wave = threadIdx.x >> 5;
  const int lane = threadIdx.x & 31;
  const int l    = lane & 15;
  const int half = lane >> 4;
  const int n0   = (blockIdx.x * 8 + wave) * 16;
  const __bf16* zcol = Z + (size_t)(n0 + l) * 64;

  float best = 3.4e38f;
  int   bi   = 0;

  for (int mt = 0; mt < 32; ++mt) {
    v8f acc = {};
#pragma unroll
    for (int kk = 0; kk < 64; kk += 32) {
      FragBF fa, fb;
      const __bf16* arow = CBB + (size_t)(mt * 16 + l) * 64 + kk + half * 8;
      fa.h[0] = *(const v8bf*)(arow);
      fa.h[1] = *(const v8bf*)(arow + 16);
      fb.h[0] = *(const v8bf*)(zcol + kk + half * 8);
      fb.h[1] = *(const v8bf*)(zcol + kk + 16 + half * 8);
      acc = __builtin_amdgcn_wmma_f32_16x16x32_bf16(
          false, fa.v, false, fb.v, (short)0, acc, false, false);
    }
#pragma unroll
    for (int r = 0; r < 8; ++r) {
      int   code = mt * 16 + half * 8 + r;
      float s    = cnorm[code] - 2.0f * acc[r];
      if (s < best) { best = s; bi = code; }
    }
  }
  float ob = __shfl_xor(best, 16, 32);
  int   oi = __shfl_xor(bi,   16, 32);
  if (ob < best || (ob == best && oi < bi)) { best = ob; bi = oi; }
  if (half == 0) idx[n0 + l] = bi;
}

__global__ void vq_gather(const int* __restrict__ idx,
                          const __bf16* __restrict__ cbb,
                          __bf16* __restrict__ Q)
{
  int i  = blockIdx.x * blockDim.x + threadIdx.x;   // 32768 * 8 threads
  int n  = i >> 3;
  int c8 = (i & 7) * 8;
  *(v8bf*)(Q + (size_t)n * 64 + c8) = *(const v8bf*)(cbb + (size_t)idx[n] * 64 + c8);
}

// ---------------------------------------------------------------------------
// First conv: Cin=1, k=3, stride 2 -> too skinny for WMMA, direct VALU.
// ---------------------------------------------------------------------------
__global__ void conv1_direct(const float* __restrict__ x,
                             const float* __restrict__ w,   // [256][1][3]
                             const float* __restrict__ bias,
                             __bf16* __restrict__ Y, int Lin, int Lout)
{
  long i = (long)blockIdx.x * blockDim.x + threadIdx.x;   // B*Lout*256
  int  o = (int)(i & 255);
  long nn = i >> 8;
  int  b = (int)(nn / Lout), t = (int)(nn % Lout);
  float s = bias[o];
#pragma unroll
  for (int k = 0; k < 3; ++k) {
    int sp = 2 * t + k - 1;
    if (sp >= 0 && sp < Lin) s += w[o * 3 + k] * x[(size_t)b * Lin + sp];
  }
  Y[((size_t)b * Lout + t) * 256 + o] = (__bf16)fmaxf(s, 0.0f);
}

// ---------------------------------------------------------------------------
// Weight repacks (fp32 [O][C][3] -> bf16 layouts), codebook bf16 + norms.
// ---------------------------------------------------------------------------
__global__ void repack_conv_w(const float* __restrict__ w,
                              __bf16* __restrict__ o, int O, int C)
{
  int i = blockIdx.x * blockDim.x + threadIdx.x;
  int total = O * C * 3;
  if (i >= total) return;
  int oo  = i / (3 * C);
  int rem = i - oo * 3 * C;
  int k   = rem / C;
  int c   = rem - k * C;
  o[i] = (__bf16)w[((size_t)oo * C + c) * 3 + k];   // out: [o][tap][c]
}

__global__ void repack_convT_w(const float* __restrict__ w,
                               __bf16* __restrict__ we,   // [O][C]   (k=1)
                               __bf16* __restrict__ wo,   // [O][2C]  (k=0,k=2)
                               int O, int C)
{
  int i = blockIdx.x * blockDim.x + threadIdx.x;
  if (i >= O * C) return;
  int oo = i / C, c = i - oo * C;
  const float* p = w + ((size_t)oo * C + c) * 3;
  we[(size_t)oo * C + c]         = (__bf16)p[1];
  wo[(size_t)oo * 2 * C + c]     = (__bf16)p[0];
  wo[(size_t)oo * 2 * C + C + c] = (__bf16)p[2];
}

__global__ void repack_cb(const float* __restrict__ cb,
                          __bf16* __restrict__ cbb, float* __restrict__ cn)
{
  int code = blockIdx.x * blockDim.x + threadIdx.x;
  if (code >= 512) return;
  float s = 0.0f;
  for (int d = 0; d < 64; ++d) {
    float v = cb[(size_t)code * 64 + d];
    s += v * v;
    cbb[(size_t)code * 64 + d] = (__bf16)v;
  }
  cn[code] = s;
}

// ---------------------------------------------------------------------------
extern "C" void kernel_launch(void* const* d_in, const int* in_sizes, int n_in,
                              void* d_out, int out_size, void* d_ws, size_t ws_size,
                              hipStream_t stream)
{
  const float* x      = (const float*)d_in[0];
  const float* ew_in  = (const float*)d_in[1];
  const float* eb_in  = (const float*)d_in[2];
  const float* ews    = (const float*)d_in[3];
  const float* ebs    = (const float*)d_in[4];
  const float* ew_out = (const float*)d_in[5];
  const float* eb_out = (const float*)d_in[6];
  const float* cb     = (const float*)d_in[7];
  const float* dw_in  = (const float*)d_in[8];
  const float* db_in  = (const float*)d_in[9];
  const float* dws    = (const float*)d_in[10];
  const float* dbs    = (const float*)d_in[11];
  const float* dw_out = (const float*)d_in[12];
  const float* db_out = (const float*)d_in[13];
  float* out = (float*)d_out;

  char*  ws  = (char*)d_ws;
  size_t off = 0;
  auto alloc = [&](size_t bytes) -> char* {
    char* p = ws + off;
    off = (off + bytes + 255) & ~(size_t)255;
    return p;
  };

  __bf16* PING = (__bf16*)alloc((size_t)16 * 16384 * 256 * 2);  // 128 MiB
  __bf16* PONG = (__bf16*)alloc((size_t)16 * 16384 * 256 * 2);  // 128 MiB
  __bf16* WAe[3];
  for (int i = 0; i < 3; ++i) WAe[i] = (__bf16*)alloc((size_t)256 * 768 * 2);
  __bf16* WAeo = (__bf16*)alloc((size_t)64 * 768 * 2);
  __bf16* WAdi = (__bf16*)alloc((size_t)256 * 192 * 2);
  __bf16* WTe[4];
  __bf16* WTo[4];
  for (int i = 0; i < 4; ++i) {
    WTe[i] = (__bf16*)alloc((size_t)256 * 256 * 2);
    WTo[i] = (__bf16*)alloc((size_t)256 * 512 * 2);
  }
  __bf16* CBB = (__bf16*)alloc((size_t)512 * 64 * 2);
  float*  CN  = (float*)alloc((size_t)512 * 4);
  int*    IDX = (int*)alloc((size_t)32768 * 4);
  __bf16* Q   = (__bf16*)alloc((size_t)16 * 2048 * 64 * 2);

  // ---- weight / codebook repacks (tiny, L2-resident afterwards) ----
  for (int i = 0; i < 3; ++i)
    repack_conv_w<<<(256 * 768 + 255) / 256, 256, 0, stream>>>(
        ews + (size_t)i * 256 * 256 * 3, WAe[i], 256, 256);
  repack_conv_w<<<(64 * 768 + 255) / 256, 256, 0, stream>>>(ew_out, WAeo, 64, 256);
  repack_conv_w<<<(256 * 192 + 255) / 256, 256, 0, stream>>>(dw_in, WAdi, 256, 64);
  for (int i = 0; i < 3; ++i)
    repack_convT_w<<<(256 * 256 + 255) / 256, 256, 0, stream>>>(
        dws + (size_t)i * 256 * 256 * 3, WTe[i], WTo[i], 256, 256);
  repack_convT_w<<<(256 * 256 + 255) / 256, 256, 0, stream>>>(dw_out, WTe[3], WTo[3], 256, 256);
  repack_cb<<<2, 256, 0, stream>>>(cb, CBB, CN);

  // ---- encoder ----
  conv1_direct<<<(16L * 16384 * 256) / 256, 256, 0, stream>>>(
      x, ew_in, eb_in, PING, 32768, 16384);                                 // A1 = PING
  conv_wmma<256, 256, 2, true, 2, 8><<<1024, 256, 0, stream>>>(
      PING, WAe[0], ebs + 0, PONG, 16384, 8192);                            // A2 = PONG
  conv_wmma<256, 256, 2, true, 2, 8><<<512, 256, 0, stream>>>(
      PONG, WAe[1], ebs + 256, PING, 8192, 4096);                           // A3 = PING
  conv_wmma<256, 256, 2, true, 2, 8><<<256, 256, 0, stream>>>(
      PING, WAe[2], ebs + 512, PONG, 4096, 2048);                           // A4 = PONG
  conv_wmma<256, 64, 1, false, 2, 2><<<256, 256, 0, stream>>>(
      PONG, WAeo, eb_out, PING, 2048, 2048);                                // Z  = PING

  // ---- VQ (st = q numerically) ----
  vq_argmin<<<256, 256, 0, stream>>>(PING, CBB, CN, IDX);
  vq_gather<<<1024, 256, 0, stream>>>(IDX, CBB, Q);

  // ---- decoder ----
  conv_wmma<64, 256, 1, true, 2, 8><<<256, 256, 0, stream>>>(
      Q, WAdi, db_in, PING, 2048, 2048);                                    // D1 = PING
  convT_wmma<false, false><<<256, 256, 0, stream>>>(PING, WTe[0], dbs + 0, PONG, 2048);
  convT_wmma<true,  false><<<256, 256, 0, stream>>>(PING, WTo[0], dbs + 0, PONG, 2048);   // D2
  convT_wmma<false, false><<<512, 256, 0, stream>>>(PONG, WTe[1], dbs + 256, PING, 4096);
  convT_wmma<true,  false><<<512, 256, 0, stream>>>(PONG, WTo[1], dbs + 256, PING, 4096); // D3
  convT_wmma<false, false><<<1024, 256, 0, stream>>>(PING, WTe[2], dbs + 512, PONG, 8192);
  convT_wmma<true,  false><<<1024, 256, 0, stream>>>(PING, WTo[2], dbs + 512, PONG, 8192);// D4
  convT_wmma<false, true><<<2048, 256, 0, stream>>>(PONG, WTe[3], db_out, out, 16384);
  convT_wmma<true,  true><<<2048, 256, 0, stream>>>(PONG, WTo[3], db_out, out, 16384);
}
